// AttentionHead_48515950575665
// MI455X (gfx1250) — matrix operand order repaired and checked by reading, not compiled, not run on hardware
//
#include <hip/hip_runtime.h>

// ---------------------------------------------------------------------------
// nGPT attention head for MI455X (gfx1250, wave32, WMMA 16x16x32 bf16 + TDM)
// B=4, T=2048, D_MODEL=D_INTERNAL=1024
// ---------------------------------------------------------------------------

#define DM   1024   // d_model == d_internal
#define TT   2048   // sequence length
#define BB   4      // batch
#define NQKV 3072   // 3 * d_internal
#define MROWS (BB*TT)      // 8192
#define CH   128    // kv chunk in attention
#define QT   16     // query rows per workgroup

typedef __attribute__((ext_vector_type(16))) __bf16 v16bf;
typedef __attribute__((ext_vector_type(8)))  __bf16 bf16x8;
typedef __attribute__((ext_vector_type(8)))  float  v8f;
typedef __attribute__((ext_vector_type(4)))  unsigned int u32x4;
typedef __attribute__((ext_vector_type(8)))  unsigned int u32x8;

// ---------------------------------------------------------------------------
// WMMA helper: D = A(16x32 bf16) * B(32x16 bf16) + C(16x16 f32)
// ---------------------------------------------------------------------------
__device__ __forceinline__ v8f wmma_bf16(v16bf a, v16bf b, v8f c) {
  return __builtin_amdgcn_wmma_f32_16x16x32_bf16(
      /*neg_a=*/false, a, /*neg_b=*/false, b,
      /*c_mod=*/(short)0, c, /*reuse_a=*/false, /*reuse_b=*/false);
}

// A-matrix fragment (16x32, bf16), row-major source with row stride `stride`.
// Lane m=lane&15 holds row m; lanes 0-15 K={k0..k0+15},
// lanes 16-31 K={k0+16..k0+31} (matches load_frag_b). Two contiguous 16B reads per lane.
__device__ __forceinline__ v16bf load_frag_a(const __bf16* base, int stride,
                                             int k0, int lane) {
  const int m  = lane & 15;
  const int kh = lane >> 4;
  const __bf16* r = base + (size_t)m * stride + k0;
  bf16x8 lo = *(const bf16x8*)(r + kh * 16);
  bf16x8 hi = *(const bf16x8*)(r + kh * 16 + 8);
  return __builtin_shufflevector(lo, hi, 0,1,2,3,4,5,6,7,8,9,10,11,12,13,14,15);
}

// B-matrix fragment (32x16, bf16): lane n=lane&15 holds col n; per-lane 16
// consecutive K values (lanes 0-15 lower K half, 16-31 upper). `rowptr` must
// already point at element (k0 + (lane>>4)*16) of the lane's source row.
__device__ __forceinline__ v16bf load_frag_b(const __bf16* rowptr) {
  bf16x8 lo = *(const bf16x8*)(rowptr);
  bf16x8 hi = *(const bf16x8*)(rowptr + 8);
  return __builtin_shufflevector(lo, hi, 0,1,2,3,4,5,6,7,8,9,10,11,12,13,14,15);
}

// ---------------------------------------------------------------------------
// Kernel 1: f32 -> bf16 conversion
// ---------------------------------------------------------------------------
__global__ void k_cvt_bf16(const float* __restrict__ in,
                           __bf16* __restrict__ out, int n) {
  int i = blockIdx.x * blockDim.x + threadIdx.x;
  if (i < n) out[i] = (__bf16)in[i];
}

// ---------------------------------------------------------------------------
// Kernel 2: QKV GEMM  C[8192,3072] = Xb[8192,1024] * Wb[3072,1024]^T   (f32 out)
// 256 threads = 8 waves as 4(M) x 2(N); wave tile 32x64; WG tile 128x128.
// Fragments loaded straight from global (working set is L2-resident),
// with global_prefetch_b8 on the next K-step.
// ---------------------------------------------------------------------------
__global__ void __launch_bounds__(256)
k_gemm_qkv(const __bf16* __restrict__ Xb, const __bf16* __restrict__ Wb,
           float* __restrict__ C) {
  const int lane = threadIdx.x & 31;
  const int wave = threadIdx.x >> 5;
  const int wm = wave >> 1;                       // 0..3
  const int wn = wave & 1;                        // 0..1
  const int Mbase = blockIdx.x * 128 + wm * 32;
  const int Nbase = blockIdx.y * 128 + wn * 64;
  const int kh = lane >> 4;
  const int ln = lane & 15;

  v8f acc[2][4];
  const v8f z8 = {0.f,0.f,0.f,0.f,0.f,0.f,0.f,0.f};
#pragma unroll
  for (int mt = 0; mt < 2; ++mt)
#pragma unroll
    for (int nt = 0; nt < 4; ++nt) acc[mt][nt] = z8;

#pragma unroll 4
  for (int k0 = 0; k0 < DM; k0 += 32) {
    if (k0 + 32 < DM) {   // prefetch next K-step (-> global_prefetch_b8)
      __builtin_prefetch(Xb + (size_t)(Mbase + ln) * DM + k0 + 32, 0, 0);
      __builtin_prefetch(Wb + (size_t)(Nbase + wn * 16 + ln) * DM + k0 + 32, 0, 0);
    }
    v16bf a0 = load_frag_a(Xb + (size_t)Mbase * DM,        DM, k0, lane);
    v16bf a1 = load_frag_a(Xb + (size_t)(Mbase + 16) * DM, DM, k0, lane);
    v16bf bf[4];
#pragma unroll
    for (int nt = 0; nt < 4; ++nt) {
      const int n = Nbase + nt * 16 + ln;
      bf[nt] = load_frag_b(Wb + (size_t)n * DM + k0 + kh * 16);
    }
#pragma unroll
    for (int nt = 0; nt < 4; ++nt) {
      acc[0][nt] = wmma_bf16(a0, bf[nt], acc[0][nt]);
      acc[1][nt] = wmma_bf16(a1, bf[nt], acc[1][nt]);
    }
  }

  // C layout per ISA: VGPR v -> row v + 8*(lane>>4), col lane&15
#pragma unroll
  for (int mt = 0; mt < 2; ++mt)
#pragma unroll
    for (int nt = 0; nt < 4; ++nt)
#pragma unroll
      for (int v = 0; v < 8; ++v) {
        const int row = Mbase + mt * 16 + v + 8 * kh;
        C[(size_t)row * NQKV + Nbase + nt * 16 + ln] = acc[mt][nt][v];
      }
}

// ---------------------------------------------------------------------------
// Kernel 3: RoPE + L2 norm + sqk scale. One 256-thread block per (b,t) row.
// Emits Qb,Kb as bf16 [b][t][d] and V transposed Vt as bf16 [b][d][t].
// ---------------------------------------------------------------------------
__global__ void __launch_bounds__(256)
k_rope_norm(const float* __restrict__ qkv, const float* __restrict__ sqk,
            __bf16* __restrict__ Qb, __bf16* __restrict__ Kb,
            __bf16* __restrict__ Vt) {
  const int m = blockIdx.x;            // b*T + t
  const int t = m & (TT - 1);
  const int b = m >> 11;               // /2048
  const int tid = threadIdx.x;
  const float* row = qkv + (size_t)m * NQKV;

  float qv[4], kv[4], vv[4];
  int idx[4];
  float ssq = 0.f, ssk = 0.f;
#pragma unroll
  for (int p = 0; p < 2; ++p) {
    const int h = tid + p * 256;                         // pair index 0..511
    const float fr = (float)t * __powf(10000.0f, -(float)h * (1.0f / 512.0f));
    const float cs = __cosf(fr), sn = __sinf(fr);
    const float q0 = row[h],        q1 = row[h + 512];
    const float k0 = row[DM + h],   k1 = row[DM + h + 512];
    const float qa = q0 * cs - q1 * sn, qb = q1 * cs + q0 * sn;
    const float ka = k0 * cs - k1 * sn, kb = k1 * cs + k0 * sn;
    qv[2*p] = qa; qv[2*p+1] = qb;
    kv[2*p] = ka; kv[2*p+1] = kb;
    idx[2*p] = h; idx[2*p+1] = h + 512;
    vv[2*p] = row[2*DM + h]; vv[2*p+1] = row[2*DM + h + 512];
    ssq += qa*qa + qb*qb;
    ssk += ka*ka + kb*kb;
  }

  __shared__ float redq[256];
  __shared__ float redk[256];
  redq[tid] = ssq; redk[tid] = ssk;
  __syncthreads();
  for (int s = 128; s > 0; s >>= 1) {
    if (tid < s) { redq[tid] += redq[tid + s]; redk[tid] += redk[tid + s]; }
    __syncthreads();
  }
  const float inq = rsqrtf(redq[0]);
  const float ink = rsqrtf(redk[0]);

#pragma unroll
  for (int e = 0; e < 4; ++e) {
    const int c = idx[e];
    const float sc = sqk[c] * 32.0f;                     // sqk * sqrt(d_model)
    Qb[(size_t)m * DM + c] = (__bf16)(qv[e] * inq * sc);
    Kb[(size_t)m * DM + c] = (__bf16)(kv[e] * ink * sc);
    Vt[((size_t)b * DM + c) * TT + t] = (__bf16)vv[e];
  }
}

// ---------------------------------------------------------------------------
// Kernel 4: causal flash attention. WG = (batch, 16 query rows), 8 waves.
// Q tile staged into LDS by the Tensor Data Mover (one descriptor, wave 0),
// S-phase: wave w computes S[:, w*16:(w+1)*16] of a 128-wide KV chunk.
// PV-phase: wave w owns output d-columns [w*128, (w+1)*128).
// ---------------------------------------------------------------------------
__global__ void __launch_bounds__(256)
k_attn(const __bf16* __restrict__ Qb, const __bf16* __restrict__ Kb,
       const __bf16* __restrict__ Vt, float* __restrict__ Out) {
  const int b   = blockIdx.x;
  const int q0  = blockIdx.y * QT;
  const int tid = threadIdx.x;
  const int lane = tid & 31;
  const int wave = tid >> 5;
  const int kh = lane >> 4;
  const int ln = lane & 15;

  __shared__ __align__(16) __bf16 sQ[QT * DM];   // 32 KB
  __shared__ float  sS[QT * CH];                 //  8 KB
  __shared__ __align__(16) __bf16 sP[QT * CH];   //  4 KB
  __shared__ float sM[QT];
  __shared__ float sL[QT];
  __shared__ float sC[QT];

  // --- TDM: one descriptor moves the whole 16x1024 bf16 Q tile to LDS ------
  // D# group 0: count=1 | lds_addr | global_addr[56:0] | type=2
  // D# group 1: data_size=2B, tensor 1024x16, tile 1024x16, dim0_stride=1024
  if (wave == 0) {
    const unsigned long long ga =
        (unsigned long long)(Qb + ((size_t)b * TT + q0) * DM);
    const unsigned lds = (unsigned)(unsigned long long)(&sQ[0]);
    u32x4 g0;
    g0[0] = 1u;                                          // count=1 (valid D#)
    g0[1] = lds;                                         // lds_addr (bytes)
    g0[2] = (unsigned)(ga & 0xFFFFFFFFu);                // global_addr[31:0]
    g0[3] = (unsigned)((ga >> 32) & 0x01FFFFFFu)         // global_addr[56:32]
          | 0x80000000u;                                 // type=2 ("image")
    u32x8 g1;
    g1[0] = 0x00010000u;                 // data_size=1 (2 bytes)
    g1[1] = (1024u << 16);               // tensor_dim0[15:0]  (bits 63:48)
    g1[2] = (16u << 16);                 // tensor_dim0[31:16]=0 | tensor_dim1=16
    g1[3] = (1024u << 16);               // tensor_dim1 hi=0 | tile_dim0=1024
    g1[4] = 16u;                         // tile_dim1=16, tile_dim2=0
    g1[5] = 1024u;                       // tensor_dim0_stride[31:0]
    g1[6] = 0u;                          // stride hi | tensor_dim1_stride lo
    g1[7] = 0u;
    asm volatile("tensor_load_to_lds %0, %1" :: "s"(g0), "s"(g1) : "memory");
    __builtin_amdgcn_s_wait_tensorcnt(0);
  }
  if (tid < QT) { sM[tid] = -1e30f; sL[tid] = 0.0f; }

  v8f accO[8];
  const v8f z8 = {0.f,0.f,0.f,0.f,0.f,0.f,0.f,0.f};
#pragma unroll
  for (int tcol = 0; tcol < 8; ++tcol) accO[tcol] = z8;
  __syncthreads();

  const int nch = q0 / CH + 1;                   // causal chunk count
  for (int j = 0; j < nch; ++j) {
    // ---- S = (Q * 32) K^T for this 128-wide chunk --------------------------
    {
      const int kv0 = j * CH + wave * 16;
      v8f acc = z8;
      const __bf16* krow = Kb + ((size_t)b * TT + kv0 + ln) * DM + kh * 16;
#pragma unroll 8
      for (int k0 = 0; k0 < DM; k0 += 32) {
        v16bf a  = load_frag_a(sQ, DM, k0, lane);
        v16bf bf = load_frag_b(krow + k0);
        acc = wmma_bf16(a, bf, acc);
      }
#pragma unroll
      for (int v = 0; v < 8; ++v) {
        const int r  = v + 8 * kh;
        const int qg = q0 + r;
        const int kg = kv0 + ln;
        float s = acc[v] * 32.0f;                // attention scale sqrt(d_model)
        if (kg > qg) s = -1e30f;                 // causal mask
        sS[r * CH + wave * 16 + ln] = s;
      }
    }
    __syncthreads();

    // ---- online softmax (one thread per query row) -------------------------
    if (tid < QT) {
      const int r = tid;
      const float mprev = sM[r];
      float mmax = mprev;
      for (int c = 0; c < CH; ++c) mmax = fmaxf(mmax, sS[r * CH + c]);
      const float corr = __expf(mprev - mmax);
      float lsum = 0.f;
      for (int c = 0; c < CH; ++c) {
        const float p = __expf(sS[r * CH + c] - mmax);
        sP[r * CH + c] = (__bf16)p;
        lsum += p;
      }
      sL[r] = sL[r] * corr + lsum;
      sM[r] = mmax;
      sC[r] = corr;
    }
    __syncthreads();

    // ---- O = O*corr + P @ V chunk ------------------------------------------
    {
#pragma unroll
      for (int tcol = 0; tcol < 8; ++tcol)
#pragma unroll
        for (int v = 0; v < 8; ++v) accO[tcol][v] *= sC[v + 8 * kh];

      const int colbase = wave * 128;
#pragma unroll
      for (int ks = 0; ks < 4; ++ks) {
        const int k0 = ks * 32;
        v16bf a = load_frag_a(sP, CH, k0, lane);
#pragma unroll
        for (int tcol = 0; tcol < 8; ++tcol) {
          const __bf16* vrow = Vt + ((size_t)b * DM + colbase + tcol * 16 + ln) * TT
                             + j * CH + k0 + kh * 16;
          v16bf bf = load_frag_b(vrow);
          accO[tcol] = wmma_bf16(a, bf, accO[tcol]);
        }
      }
    }
    __syncthreads();
  }

  // ---- epilogue: O /= l, store f32 -----------------------------------------
  {
    const int colbase = wave * 128;
#pragma unroll
    for (int tcol = 0; tcol < 8; ++tcol)
#pragma unroll
      for (int v = 0; v < 8; ++v) {
        const int r = v + 8 * kh;
        const float o = accO[tcol][v] / sL[r];
        Out[((size_t)b * TT + q0 + r) * DM + colbase + tcol * 16 + ln] = o;
      }
  }
}

// ---------------------------------------------------------------------------
// Host launcher
// ---------------------------------------------------------------------------
extern "C" void kernel_launch(void* const* d_in, const int* in_sizes, int n_in,
                              void* d_out, int out_size, void* d_ws, size_t ws_size,
                              hipStream_t stream) {
  (void)in_sizes; (void)n_in; (void)out_size; (void)ws_size;

  const float* x   = (const float*)d_in[0];   // [B,T,DM]   f32
  const float* w   = (const float*)d_in[1];   // [3*DM,DM]  f32
  const float* sqk = (const float*)d_in[2];   // [DM]       f32
  float* out = (float*)d_out;                 // [B,T,DM]   f32

  char* ws = (char*)d_ws;
  const size_t nX  = (size_t)MROWS * DM;       // 8,388,608
  const size_t nW  = (size_t)NQKV * DM;        // 3,145,728
  __bf16* Xb  = (__bf16*)(ws);                                   size_t off = nX * 2;
  __bf16* Wb  = (__bf16*)(ws + off);                             off += nW * 2;
  float*  qkv = (float*) (ws + off);                             off += (size_t)MROWS * NQKV * 4;
  __bf16* Qb  = (__bf16*)(ws + off);                             off += nX * 2;
  __bf16* Kb  = (__bf16*)(ws + off);                             off += nX * 2;
  __bf16* Vt  = (__bf16*)(ws + off);

  k_cvt_bf16<<<(int)((nX + 255) / 256), 256, 0, stream>>>(x, Xb, (int)nX);
  k_cvt_bf16<<<(int)((nW + 255) / 256), 256, 0, stream>>>(w, Wb, (int)nW);

  k_gemm_qkv<<<dim3(MROWS / 128, NQKV / 128), 256, 0, stream>>>(Xb, Wb, qkv);

  k_rope_norm<<<MROWS, 256, 0, stream>>>(qkv, sqk, Qb, Kb, Vt);

  k_attn<<<dim3(BB, TT / QT), 256, 0, stream>>>(Qb, Kb, Vt, out);
}
